// LRPEMultiHeadAttention_76381698392649
// MI455X (gfx1250) — compile-verified, matrix operand order checked by reading, not run on hardware
//
#include <hip/hip_runtime.h>
#include <hip/hip_bf16.h>
#include <math.h>

// ---------------------------------------------------------------------------
// LRPE multi-head attention for MI455X (gfx1250, wave32, WMMA bf16 K=32 == Dh)
// Data staged with CDNA5 async direct-to-LDS copies (ASYNCcnt path).
// ---------------------------------------------------------------------------

#define B_  4
#define N_  2048
#define M_  2048
#define C_  256
#define H_  8
#define DH_ 32
#define R_  128

typedef __attribute__((ext_vector_type(16))) __bf16 v16bf;
typedef __attribute__((ext_vector_type(8)))  __bf16 v8bf;
typedef __attribute__((ext_vector_type(8)))  float  v8f;

// K-index held by element e of a 16x32 bf16 A-fragment (or 32x16 B-fragment),
// for the lane-half g = lane/16.  (ISA 7.12.2, 16-bit A-matrix 16x32 layout.)
__device__ __forceinline__ int kmap(int e, int g) {
    int half = e >> 3;        // VGPR 0-3 vs 4-7
    int j    = (e & 7) >> 1;  // VGPR within group
    int p    = e & 1;         // low/high half of dword
    return half * 16 + j * 2 + g * 8 + p;
}

__device__ __forceinline__ v8f wmma_bf16(v16bf a, v16bf b, v8f c) {
    return __builtin_amdgcn_wmma_f32_16x16x32_bf16(
        /*neg_a=*/false, a, /*neg_b=*/false, b,
        /*c_mod=*/(short)0, c, /*reuse_a=*/false, /*reuse_b=*/false);
}

// CDNA5 async copy: 16B global -> LDS, no VGPR round-trip, tracked by ASYNCcnt.
// lds_off = low 32 bits of the generic LDS address (= LDS byte offset).
__device__ __forceinline__ void async_copy_b128(unsigned lds_off, const void* src) {
    asm volatile("global_load_async_to_lds_b128 %0, %1, off"
                 :: "v"(lds_off), "v"((unsigned long long)(size_t)src)
                 : "memory");
}
__device__ __forceinline__ void wait_asynccnt0() {
    asm volatile("s_wait_asynccnt 0" ::: "memory");
}
__device__ __forceinline__ unsigned lds_offset(const void* p) {
    return (unsigned)(size_t)p;   // generic LDS addr low 32 bits == LDS offset
}

// ---------------------------------------------------------------------------
// Kernel 1: out[row, c] = X[row, :] @ W[:, c] + bias[c]   (f32 in, bf16 out)
// block = 128 threads (4 waves); block tile = 64 rows x 64 cols; K step = 32.
// Each wave: 16 rows x 64 cols -> 4 WMMAs per K-step off one A-fragment.
// ---------------------------------------------------------------------------
__global__ __launch_bounds__(128)
void lrpe_proj_kernel(const float* __restrict__ X, const float* __restrict__ W,
                      const float* __restrict__ bias, __bf16* __restrict__ out)
{
    __shared__ float lds_a[64 * 32];   // 8 KB
    __shared__ float lds_b[32 * 64];   // 8 KB

    const int t    = threadIdx.x;
    const int w    = t >> 5;
    const int lane = t & 31;
    const int g    = lane >> 4;
    const int ln   = lane & 15;

    const int rowBase = blockIdx.x * 64;
    const int colBase = blockIdx.y * 64;

    const unsigned a_base = lds_offset(&lds_a[0]);
    const unsigned b_base = lds_offset(&lds_b[0]);

    v8f acc[4] = {{}, {}, {}, {}};

    for (int k0 = 0; k0 < C_; k0 += 32) {
        // async-stage A (64x32 f32) and B (32x64 f32): 512 x 16B units each
        #pragma unroll
        for (int i = 0; i < 4; ++i) {
            int u   = i * 128 + t;
            int row = u >> 3;            // 8 units per 32-float row
            int c4  = (u & 7) * 4;
            async_copy_b128(a_base + u * 16,
                            X + (size_t)(rowBase + row) * C_ + k0 + c4);
        }
        #pragma unroll
        for (int i = 0; i < 4; ++i) {
            int u  = i * 128 + t;
            int kk = u >> 4;             // 16 units per 64-float row
            int c4 = (u & 15) * 4;
            async_copy_b128(b_base + u * 16,
                            W + (size_t)(k0 + kk) * C_ + colBase + c4);
        }
        wait_asynccnt0();
        __syncthreads();

        v16bf af;
        #pragma unroll
        for (int e = 0; e < 16; ++e)
            af[e] = (__bf16)lds_a[(w * 16 + ln) * 32 + kmap(e, g)];

        #pragma unroll
        for (int j = 0; j < 4; ++j) {
            v16bf bf;
            #pragma unroll
            for (int e = 0; e < 16; ++e)
                bf[e] = (__bf16)lds_b[kmap(e, g) * 64 + j * 16 + ln];
            acc[j] = wmma_bf16(af, bf, acc[j]);
        }
        __syncthreads();
    }

    #pragma unroll
    for (int j = 0; j < 4; ++j) {
        const float bv = bias[colBase + j * 16 + ln];
        #pragma unroll
        for (int r = 0; r < 8; ++r) {
            int row = rowBase + w * 16 + r + 8 * g;
            out[(size_t)row * C_ + colBase + j * 16 + ln] = (__bf16)(acc[j][r] + bv);
        }
    }
}

// ---------------------------------------------------------------------------
// Kernel 2: p_all[b,h,n,r] = sum_d q[b,n,h*32+d] * rpe_bank[r, h*32+d]
// one wave computes a 16 x 128 strip: 8 WMMAs reusing a single Q fragment.
// ---------------------------------------------------------------------------
__global__ __launch_bounds__(32)
void lrpe_rpe_scores_kernel(const __bf16* __restrict__ qb,
                            const float* __restrict__ rpe_bank,
                            float* __restrict__ p_all)
{
    const int lane = threadIdx.x;
    const int g    = lane >> 4;
    const int ln   = lane & 15;
    const int b    = blockIdx.z / H_;
    const int h    = blockIdx.z % H_;
    const int nb   = blockIdx.x * 16;

    v16bf af;
    #pragma unroll
    for (int e = 0; e < 16; ++e)
        af[e] = qb[(size_t)(b * N_ + nb + ln) * C_ + h * DH_ + kmap(e, g)];

    #pragma unroll
    for (int rt = 0; rt < 8; ++rt) {
        const int rb = rt * 16;
        v16bf bf;
        #pragma unroll
        for (int e = 0; e < 16; ++e)
            bf[e] = (__bf16)rpe_bank[(size_t)(rb + ln) * C_ + h * DH_ + kmap(e, g)];
        v8f acc = {};
        acc = wmma_bf16(af, bf, acc);
        #pragma unroll
        for (int r = 0; r < 8; ++r) {
            int n = nb + r + 8 * g;
            p_all[((size_t)(b * H_ + h) * N_ + n) * R_ + rb + ln] = acc[r];
        }
    }
}

// ---------------------------------------------------------------------------
// Kernel 3: flash attention with RPE gather.
// block = 128 threads (4 waves), 64 query rows; m-chunks of 32.
// K/V chunks and per-wave P rows staged via async direct-to-LDS copies.
// ---------------------------------------------------------------------------
__global__ __launch_bounds__(128)
void lrpe_flash_attn_kernel(const __bf16* __restrict__ qb,
                            const __bf16* __restrict__ kb,
                            const __bf16* __restrict__ vb,
                            const float* __restrict__ p_all,
                            const int*   __restrict__ rpe_idx,
                            const float* __restrict__ key_masks,
                            float* __restrict__ out)
{
    __shared__ __bf16 lds_k[32 * 32];          // 2 KB
    __shared__ __bf16 lds_v[32 * 32];          // 2 KB
    __shared__ __bf16 lds_p[4][16 * 32];       // 4 KB: per-wave P tile for PV
    __shared__ float  lds_prow[4][16 * R_];    // 32 KB: per-wave cached RPE rows

    const int t    = threadIdx.x;
    const int w    = t >> 5;
    const int lane = t & 31;
    const int g    = lane >> 4;
    const int ln   = lane & 15;
    const int b    = blockIdx.z;
    const int h    = blockIdx.y;
    const int nbase = blockIdx.x * 64 + w * 16;

    // async-stage this wave's 16 rows of p_all (16 x 128 f32 = 512 x 16B):
    // read once from HBM, gathered 2048x each from LDS.
    {
        const unsigned prow_base = lds_offset(&lds_prow[w][0]);
        #pragma unroll
        for (int i = 0; i < 16; ++i) {
            int u   = i * 32 + lane;     // unit in this wave's 512-unit region
            int row = u >> 5;            // 32 units per 128-float row
            int c4  = (u & 31) * 4;
            async_copy_b128(prow_base + u * 16,
                p_all + ((size_t)(b * H_ + h) * N_ + nbase + row) * R_ + c4);
        }
    }

    // Q fragment (16 rows x 32 d), loaded once
    v16bf qf;
    #pragma unroll
    for (int e = 0; e < 16; ++e)
        qf[e] = qb[(size_t)(b * N_ + nbase + ln) * C_ + h * DH_ + kmap(e, g)];

    v8f o0 = {}, o1 = {};
    float mrun[8], lrun[8];
    #pragma unroll
    for (int r = 0; r < 8; ++r) { mrun[r] = -INFINITY; lrun[r] = 0.f; }

    const float scale = 0.17677669529663687f;  // 1/sqrt(32)
    const unsigned k_base = lds_offset(&lds_k[0]);
    const unsigned v_base = lds_offset(&lds_v[0]);

    wait_asynccnt0();   // P rows resident (wave-local region, no barrier needed)

    for (int mb = 0; mb < M_; mb += 32) {
        // async-stage K, V chunk (32 m x 32 d bf16): 16B per thread each
        {
            int m  = t >> 2;
            int d0 = (t & 3) * 8;
            size_t src = (size_t)(b * M_ + mb + m) * C_ + h * DH_ + d0;
            unsigned loff = (unsigned)(m * 32 + d0) * 2;
            async_copy_b128(k_base + loff, kb + src);
            async_copy_b128(v_base + loff, vb + src);
        }
        wait_asynccnt0();
        __syncthreads();

        // stream-ahead prefetch of the next rpe_indices block
        if (mb + 32 < M_)
            __builtin_prefetch(&rpe_idx[(size_t)(b * N_ + nbase + ln) * M_ + mb + 32], 0, 1);

        // S = Q K^T  (two 16x16 tiles), + gathered RPE + mask, scaled
        float s[2][8];
        #pragma unroll
        for (int half = 0; half < 2; ++half) {
            const int mi = half * 16;
            v16bf kf;
            #pragma unroll
            for (int e = 0; e < 16; ++e)
                kf[e] = lds_k[(mi + ln) * 32 + kmap(e, g)];
            v8f sc = {};
            sc = wmma_bf16(qf, kf, sc);

            const int m  = mb + mi + ln;
            const float km = key_masks[b * M_ + m];
            #pragma unroll
            for (int r = 0; r < 8; ++r) {
                int n   = nbase + r + 8 * g;
                int idx = rpe_idx[(size_t)(b * N_ + n) * M_ + m];
                float p = lds_prow[w][(r + 8 * g) * R_ + idx];
                s[half][r] = (sc[r] + p) * scale + km;
            }
        }

        // online softmax; rows live across the 16 lanes of each half (xor<=8
        // shuffles stay inside a half on wave32)
        #pragma unroll
        for (int r = 0; r < 8; ++r) {
            float mx = fmaxf(s[0][r], s[1][r]);
            #pragma unroll
            for (int off = 8; off > 0; off >>= 1)
                mx = fmaxf(mx, __shfl_xor(mx, off, 32));
            float mnew  = fmaxf(mrun[r], mx);
            float alpha = __expf(mrun[r] - mnew);
            float p0 = __expf(s[0][r] - mnew);
            float p1 = __expf(s[1][r] - mnew);
            float lsum = p0 + p1;
            #pragma unroll
            for (int off = 8; off > 0; off >>= 1)
                lsum += __shfl_xor(lsum, off, 32);
            lrun[r] = lrun[r] * alpha + lsum;
            mrun[r] = mnew;
            o0[r] *= alpha;
            o1[r] *= alpha;
            lds_p[w][(r + 8 * g) * 32 + ln]      = (__bf16)p0;
            lds_p[w][(r + 8 * g) * 32 + 16 + ln] = (__bf16)p1;
        }

        // O += P (16n x 32m) @ V (32m x 32d): two WMMAs for the two d-halves
        v16bf pf, vf0, vf1;
        #pragma unroll
        for (int e = 0; e < 16; ++e) {
            int k = kmap(e, g);
            pf[e]  = lds_p[w][ln * 32 + k];
            vf0[e] = lds_v[k * 32 + ln];
            vf1[e] = lds_v[k * 32 + 16 + ln];
        }
        o0 = wmma_bf16(pf, vf0, o0);
        o1 = wmma_bf16(pf, vf1, o1);

        __syncthreads();
    }

    #pragma unroll
    for (int r = 0; r < 8; ++r) {
        int n = nbase + r + 8 * g;
        float inv = 1.f / lrun[r];
        size_t o = (size_t)(b * N_ + n) * C_ + h * DH_;
        out[o + ln]      = o0[r] * inv;
        out[o + 16 + ln] = o1[r] * inv;
    }
}

// ---------------------------------------------------------------------------
extern "C" void kernel_launch(void* const* d_in, const int* in_sizes, int n_in,
                              void* d_out, int out_size, void* d_ws, size_t ws_size,
                              hipStream_t stream) {
    const float* input_q   = (const float*)d_in[0];
    const float* input_k   = (const float*)d_in[1];
    const float* input_v   = (const float*)d_in[2];
    const int*   rpe_idx   = (const int*)  d_in[3];
    const float* key_masks = (const float*)d_in[4];
    const float* Wq        = (const float*)d_in[5];
    const float* bq        = (const float*)d_in[6];
    const float* Wk        = (const float*)d_in[7];
    const float* bk        = (const float*)d_in[8];
    const float* Wv        = (const float*)d_in[9];
    const float* bv        = (const float*)d_in[10];
    const float* rpe_bank  = (const float*)d_in[11];
    float* out = (float*)d_out;

    // workspace layout
    char* ws = (char*)d_ws;
    const size_t proj_elems = (size_t)B_ * N_ * C_;        // 2 Mi elems
    __bf16* qb    = (__bf16*)(ws);
    __bf16* kb    = (__bf16*)(ws + proj_elems * 2);
    __bf16* vbuf  = (__bf16*)(ws + proj_elems * 4);
    float*  p_all = (float*) (ws + proj_elems * 6);        // B*H*N*R f32 = 32 MiB

    const dim3 projGrid((B_ * N_) / 64, C_ / 64, 1);
    lrpe_proj_kernel<<<projGrid, 128, 0, stream>>>(input_q, Wq, bq, qb);
    lrpe_proj_kernel<<<projGrid, 128, 0, stream>>>(input_k, Wk, bk, kb);
    lrpe_proj_kernel<<<projGrid, 128, 0, stream>>>(input_v, Wv, bv, vbuf);

    lrpe_rpe_scores_kernel<<<dim3(N_ / 16, 1, B_ * H_), 32, 0, stream>>>(
        qb, rpe_bank, p_all);

    lrpe_flash_attn_kernel<<<dim3(N_ / 64, H_, B_), 128, 0, stream>>>(
        qb, kb, vbuf, p_all, rpe_idx, key_masks, out);
}
